// PKModel_23776938950703
// MI455X (gfx1250) — matrix operand order repaired and checked by reading, not compiled, not run on hardware
//
#include <hip/hip_runtime.h>
#include <hip/hip_bf16.h>
#include <stdint.h>

// ---------------------------------------------------------------------------
// PK GRU scan, MI455X (gfx1250, wave32, WMMA).
//
//   * W_ih / W_hh converted once to bf16 AND pre-swizzled into the native
//     WMMA A-fragment layout: each 16-row x 32-K tile is a contiguous 1KB
//     block (lane-major, 32B per lane). Weight streaming is then perfectly
//     coalesced contiguous wave reads, L2-resident (50 MB << 192 MB L2).
//   * Persistent kernel: 128 WGs x 192 threads (6 waves). WG t owns hidden
//     rows [16t,16t+16). Wave w computes one (matrix, gate) GEMV tile via
//     64x V_WMMA_F32_16X16X32_BF16 with the x/h vector broadcast into all
//     16 B columns (layout-robust readout from any column).
//   * K-loop unrolled x4 with two independent WMMA accumulator chains +
//     global_prefetch 8KB ahead.
//   * Two sense-reversing grid barriers per step (s_sleep in spin).
// ---------------------------------------------------------------------------

typedef __attribute__((ext_vector_type(16))) __bf16 bf16x16;
typedef __attribute__((ext_vector_type(8)))  __bf16 bf16x8;
typedef __attribute__((ext_vector_type(8)))  float  v8f;

#define H      2048
#define G3     6144        // 3*H gate rows
#define LSEQ   256
#define NSTEPS 255
#define NB     128         // workgroups (each owns 16 hidden indices)
#define TPB    192         // 6 waves of 32
#define KCH    64          // K chunks of 32 per GEMV row-tile
#define CHELEM 512         // bf16 elements per swizzled 1KB tile-chunk

// ---------------------------------------------------------------- grid barrier
__device__ __forceinline__ void grid_barrier(int* cnt, int* gen) {
    __syncthreads();
    if (threadIdx.x == 0) {
        __threadfence();                                   // release prior writes
        int g = __hip_atomic_load(gen, __ATOMIC_RELAXED, __HIP_MEMORY_SCOPE_AGENT);
        int v = __hip_atomic_fetch_add(cnt, 1, __ATOMIC_ACQ_REL, __HIP_MEMORY_SCOPE_AGENT);
        if (v == NB - 1) {
            __hip_atomic_store(cnt, 0, __ATOMIC_RELAXED, __HIP_MEMORY_SCOPE_AGENT);
            __hip_atomic_fetch_add(gen, 1, __ATOMIC_RELEASE, __HIP_MEMORY_SCOPE_AGENT);
        } else {
            while (__hip_atomic_load(gen, __ATOMIC_ACQUIRE, __HIP_MEMORY_SCOPE_AGENT) == g) {
                __builtin_amdgcn_s_sleep(1);
            }
        }
        __threadfence();                                   // acquire: invalidate near caches
    }
    __syncthreads();
}

__device__ __forceinline__ float fast_sigmoid(float v) {
    // 1 / (1 + e^-v) with v_rcp instead of full IEEE divide
    return __builtin_amdgcn_rcpf(1.0f + __expf(-v));
}

// ---------------------------------------------- f32 -> bf16 + WMMA-A swizzle
// Output layout: tile (16 rows) -> 64 chunks of 1KB. Chunk c holds the A
// fragment for K=[32c,32c+32): lane L's 32 bytes at offset L*32 are
//   part0: row (L&15), k = 32c + 8*(L>>4) + j      (j=0..7)
//   part1: row (L&15), k = 32c + 16 + 8*(L>>4) + j
__global__ void conv_swizzle_kernel(const float* __restrict__ src,
                                    __bf16* __restrict__ dst, int n) {
    int i = blockIdx.x * blockDim.x + threadIdx.x;
    int stride = gridDim.x * blockDim.x;
    for (; i < n; i += stride) {
        int e    = i & (CHELEM - 1);      // element within 1KB chunk
        int cid  = i >> 9;                // global chunk id
        int c    = cid & (KCH - 1);       // k-chunk within tile
        int tile = cid >> 6;              // 16-row tile (0..383)
        int lane = e >> 4;
        int w    = e & 15;
        int part = w >> 3;
        int j    = w & 7;
        int m    = lane & 15;
        int hh   = lane >> 4;
        int k    = 32 * c + 16 * part + 8 * hh + j;
        int row  = tile * 16 + m;
        dst[i] = (__bf16)src[(size_t)row * H + k];
    }
}

// ----------------------------------------------------------------- init state
__global__ void init_state_kernel(const float* __restrict__ timepoints,
                                  const float* __restrict__ pk_data,
                                  const int*   __restrict__ emb_patient,
                                  const float* __restrict__ drug_route,
                                  const float* __restrict__ dose,
                                  const float* __restrict__ smiles,
                                  const float* __restrict__ emb_table,
                                  const float* __restrict__ W_pre,
                                  const float* __restrict__ b_pre,
                                  float*  __restrict__ d_out,
                                  __bf16* __restrict__ xbuf,
                                  __bf16* __restrict__ hbuf,   // 2*H, parity 0 first
                                  float*  __restrict__ hf32,
                                  float*  __restrict__ pred_acc,
                                  int*    __restrict__ bar) {
    int tid = threadIdx.x;
    for (int i = tid; i < H; i += TPB) {
        float h = smiles[i];
        hf32[i] = h;
        hbuf[i] = (__bf16)h;             // parity-0 slot
    }
    if (tid < LSEQ) pred_acc[tid] = 0.0f;
    if (tid < 2)    bar[tid] = 0;
    if (tid == 0)   d_out[0] = pk_data[0];

    // x_0 = W_pre @ x7_0 + b_pre
    int ep = emb_patient[0];
    float x7[7];
    x7[0] = pk_data[0];
    x7[1] = timepoints[0];
    x7[2] = dose[0];
    x7[3] = emb_table[ep * 3 + 0];
    x7[4] = emb_table[ep * 3 + 1];
    x7[5] = emb_table[ep * 3 + 2];
    x7[6] = drug_route[0];
    for (int d = tid; d < H; d += TPB) {
        float a = b_pre[d];
        #pragma unroll
        for (int c = 0; c < 7; c++) a += W_pre[d * 7 + c] * x7[c];
        xbuf[d] = (__bf16)a;
    }
}

// --------------------------------------------------------- persistent GRU scan
__global__ __launch_bounds__(TPB) void gru_scan_kernel(
        const float* __restrict__ timepoints,
        const float* __restrict__ pk_data,
        const int*   __restrict__ tf_mask,
        const int*   __restrict__ emb_patient,
        const float* __restrict__ drug_route,
        const float* __restrict__ dose,
        const float* __restrict__ emb_table,
        const float* __restrict__ W_pre,
        const float* __restrict__ b_pre,
        const float* __restrict__ b_ih,
        const float* __restrict__ b_hh,
        const float* __restrict__ W_out,
        const float* __restrict__ b_out,
        const __bf16* __restrict__ WihS,    // swizzled bf16 [384 tiles][64KB... 64 chunks * 1KB]
        const __bf16* __restrict__ WhhS,
        __bf16* __restrict__ xbuf,          // bf16 [2048]
        __bf16* __restrict__ hbuf,          // bf16 [2][2048]
        float*  __restrict__ hf32,          // f32  [2048]
        float*  __restrict__ pred_acc,      // f32  [256]
        int*    __restrict__ bar,           // {cnt, gen}
        float*  __restrict__ d_out) {

    __shared__ __attribute__((aligned(32))) __bf16 sX[H];
    __shared__ __attribute__((aligned(32))) __bf16 sH[H];
    __shared__ float sG[6][16];
    __shared__ float sPred;

    const int tid  = threadIdx.x;
    const int wave = tid >> 5;
    const int lane = tid & 31;
    const int m    = lane & 15;
    const int half = lane >> 4;
    const int t    = blockIdx.x;       // hidden tile id: rows 16t..16t+15

    // add_info (uniform across grid)
    const int ep  = emb_patient[0];
    const float ai0 = emb_table[ep * 3 + 0];
    const float ai1 = emb_table[ep * 3 + 1];
    const float ai2 = emb_table[ep * 3 + 2];
    const float ai3 = drug_route[0];
    const float dose0 = dose[0];

    const int   mat   = wave / 3;      // 0: W_ih@x, 1: W_hh@h   (waves 0..5)
    const int   gate  = wave % 3;      // r, z, n
    const int   gtile = gate * (H / 16) + t;                 // global 16-row tile
    const __bf16* wbase = (mat ? WhhS : WihS)
                        + (size_t)gtile * (KCH * CHELEM) + lane * 16;

    for (int s = 0; s < NSTEPS; ++s) {
        // ---- stage x_s and h_s into LDS (4KB each) ----
        {
            const uint32_t* gx = (const uint32_t*)xbuf;
            const uint32_t* gh = (const uint32_t*)(hbuf + (size_t)(s & 1) * H);
            uint32_t* lx = (uint32_t*)sX;
            uint32_t* lh = (uint32_t*)sH;
            for (int i = tid; i < H / 2; i += TPB) {
                lx[i] = gx[i];
                lh[i] = gh[i];
            }
        }
        __syncthreads();

        // ---- 6 GEMV tiles via WMMA ----
        if (wave < 6) {
            const __bf16* vec = mat ? (const __bf16*)sH : (const __bf16*)sX;
            const __bf16* wp  = wbase;
            v8f acc0 = {0.f, 0.f, 0.f, 0.f, 0.f, 0.f, 0.f, 0.f};
            v8f acc1 = {0.f, 0.f, 0.f, 0.f, 0.f, 0.f, 0.f, 0.f};
            for (int c = 0; c < KCH; c += 4) {
                __builtin_prefetch(wp + 8 * CHELEM, 0, 1);   // 8KB ahead in weight stream
                #pragma unroll
                for (int u = 0; u < 4; ++u) {
                    union { bf16x16 v; bf16x8 p[2]; } A, B;
                    // swizzled A fragment: contiguous 32B per lane
                    A.p[0] = *(const bf16x8*)(wp);
                    A.p[1] = *(const bf16x8*)(wp + 8);
                    wp += CHELEM;
                    // B: vector broadcast into all 16 columns
                    const int vb = (c + u) * 32 + 8 * half;
                    B.p[0] = *(const bf16x8*)(vec + vb);
                    B.p[1] = *(const bf16x8*)(vec + vb + 16);
                    if (u & 1)
                        acc1 = __builtin_amdgcn_wmma_f32_16x16x32_bf16(
                                false, A.v, false, B.v, (short)0, acc1, false, false);
                    else
                        acc0 = __builtin_amdgcn_wmma_f32_16x16x32_bf16(
                                false, A.v, false, B.v, (short)0, acc0, false, false);
                }
            }
            #pragma unroll
            for (int i = 0; i < 8; i++) acc0[i] += acc1[i];
            // every column holds the GEMV result; lanes 0 / 16 carry rows 0..7 / 8..15
            if (m == 0) {
                const int rb = half * 8;
                #pragma unroll
                for (int i = 0; i < 8; i++) sG[wave][rb + i] = acc0[i];
            }
        }
        __syncthreads();

        // ---- gate math for this WG's 16 hidden indices ----
        if (tid < 16) {
            const int j = t * 16 + tid;
            float ir  = sG[0][tid] + b_ih[j];
            float iz  = sG[1][tid] + b_ih[H + j];
            float in_ = sG[2][tid] + b_ih[2 * H + j];
            float hr  = sG[3][tid] + b_hh[j];
            float hz  = sG[4][tid] + b_hh[H + j];
            float hn  = sG[5][tid] + b_hh[2 * H + j];
            float r = fast_sigmoid(ir + hr);
            float z = fast_sigmoid(iz + hz);
            float n = tanhf(in_ + r * hn);
            float hold = hf32[j];
            float hnew = (1.0f - z) * n + z * hold;
            hf32[j] = hnew;
            hbuf[(size_t)((s + 1) & 1) * H + j] = (__bf16)hnew;
            float p = W_out[j] * hnew;
            #pragma unroll
            for (int off = 8; off >= 1; off >>= 1) p += __shfl_down(p, off, 16);
            if (tid == 0) atomicAdd(&pred_acc[s], p);
        }

        grid_barrier(&bar[0], &bar[1]);   // pred_acc + h fully visible

        // ---- finalize pred (one contended load per WG), broadcast via LDS ----
        if (tid == 0) {
            sPred = __hip_atomic_load(&pred_acc[s], __ATOMIC_ACQUIRE,
                                      __HIP_MEMORY_SCOPE_AGENT) + b_out[0];
        }
        __syncthreads();
        const float pred = sPred;
        if (t == 0 && tid == 0) d_out[s + 1] = pred;

        // ---- teacher forcing + x_{s+1} (this WG's 16 entries) ----
        if (s < NSTEPS - 1 && tid < 16) {
            float pk_next = (tf_mask[s + 1] == 1) ? pred : pk_data[s + 1];
            float x7[7] = {pk_next, timepoints[s + 1], dose0, ai0, ai1, ai2, ai3};
            const int j = t * 16 + tid;
            float a = b_pre[j];
            #pragma unroll
            for (int c = 0; c < 7; c++) a += W_pre[j * 7 + c] * x7[c];
            xbuf[j] = (__bf16)a;
        }

        grid_barrier(&bar[0], &bar[1]);   // x_{s+1} visible before next step
    }
}

// ---------------------------------------------------------------------------
extern "C" void kernel_launch(void* const* d_in, const int* in_sizes, int n_in,
                              void* d_out, int out_size, void* d_ws, size_t ws_size,
                              hipStream_t stream) {
    const float* timepoints = (const float*)d_in[0];
    const float* pk_data    = (const float*)d_in[1];
    // d_in[2] = input_len (unused; L fixed)
    const int*   emb_pat    = (const int*)  d_in[3];
    const float* drug_route = (const float*)d_in[4];
    const float* dose       = (const float*)d_in[5];
    const float* smiles     = (const float*)d_in[6];
    const int*   tf_mask    = (const int*)  d_in[7];
    const float* emb_table  = (const float*)d_in[8];
    const float* W_pre      = (const float*)d_in[9];
    const float* b_pre      = (const float*)d_in[10];
    const float* W_ih       = (const float*)d_in[11];
    const float* W_hh       = (const float*)d_in[12];
    const float* b_ih       = (const float*)d_in[13];
    const float* b_hh       = (const float*)d_in[14];
    const float* W_out      = (const float*)d_in[15];
    const float* b_out      = (const float*)d_in[16];
    float* out = (float*)d_out;

    // ---- workspace carve-out (256B aligned slots) ----
    char* ws = (char*)d_ws;
    size_t off = 0;
    auto take = [&](size_t bytes) -> void* {
        void* p = ws + off;
        off = (off + bytes + 255) & ~(size_t)255;
        return p;
    };
    __bf16* WihS = (__bf16*)take((size_t)G3 * H * sizeof(__bf16)); // 25.2 MB swizzled
    __bf16* WhhS = (__bf16*)take((size_t)G3 * H * sizeof(__bf16)); // 25.2 MB swizzled
    __bf16* xbuf = (__bf16*)take((size_t)H * sizeof(__bf16));
    __bf16* hbuf = (__bf16*)take((size_t)2 * H * sizeof(__bf16));
    float*  hf32 = (float*) take((size_t)H * sizeof(float));
    float*  pacc = (float*) take((size_t)LSEQ * sizeof(float));
    int*    bar  = (int*)   take(256);
    (void)ws_size; (void)n_in; (void)in_sizes; (void)out_size;

    // 1) weights -> bf16, pre-swizzled into WMMA A-fragment layout
    conv_swizzle_kernel<<<4096, 256, 0, stream>>>(W_ih, WihS, G3 * H);
    conv_swizzle_kernel<<<4096, 256, 0, stream>>>(W_hh, WhhS, G3 * H);

    // 2) state init (h0, x0, barrier/accumulator reset, out[0])
    init_state_kernel<<<1, TPB, 0, stream>>>(timepoints, pk_data, emb_pat,
                                             drug_route, dose, smiles, emb_table,
                                             W_pre, b_pre, out, xbuf, hbuf,
                                             hf32, pacc, bar);

    // 3) persistent 255-step GRU scan
    gru_scan_kernel<<<NB, TPB, 0, stream>>>(timepoints, pk_data, tf_mask,
                                            emb_pat, drug_route, dose, emb_table,
                                            W_pre, b_pre, b_ih, b_hh, W_out, b_out,
                                            WihS, WhhS, xbuf, hbuf, hf32,
                                            pacc, bar, out);
}
